// HybridKernelMethod_65481071406148
// MI455X (gfx1250) — compile-verified, matrix operand order checked by reading, not run on hardware
//
#include <hip/hip_runtime.h>

// ---- types ----
typedef __bf16 bf16_t;
typedef __attribute__((ext_vector_type(4)))  __bf16 bf16x4;
typedef __attribute__((ext_vector_type(8)))  __bf16 bf16x8;
typedef __attribute__((ext_vector_type(16))) __bf16 bf16x16;
typedef __attribute__((ext_vector_type(8)))  float  floatx8;

#define NROWS 8192
#define DIM   512
#define TILE_M 256         // workgroup output tile (rows)
#define TILE_N 128         // workgroup output tile (cols)
#define KC    32           // K per WMMA step
#define LDS_STRIDE 40      // halves per LDS row (32 + 8 pad -> avoid bank conflicts)

// -------------------------------------------------------------------------
// Pass 1: f32 -> bf16 conversion + row squared-norms.
// One wave (32 lanes) per row; each lane handles 4 float4's (16 elems).
// -------------------------------------------------------------------------
__global__ __launch_bounds__(256) void rbf_convert(
    const float* __restrict__ a, const float* __restrict__ b,
    bf16_t* __restrict__ abf, bf16_t* __restrict__ bbf,
    float* __restrict__ asq, float* __restrict__ bsq)
{
  int row  = blockIdx.x * 8 + (threadIdx.x >> 5);   // 0 .. 16383
  int lane = threadIdx.x & 31;

  const float* src; bf16_t* dst; float* sq; int r;
  if (row < NROWS) { src = a; dst = abf; sq = asq; r = row; }
  else             { src = b; dst = bbf; sq = bsq; r = row - NROWS; }

  const float4* s4 = (const float4*)(src + (size_t)r * DIM);
  bf16x4*       d4 = (bf16x4*)(dst + (size_t)r * DIM);

  float sum = 0.0f;
#pragma unroll
  for (int j = 0; j < 4; ++j) {
    float4 v = s4[lane + 32 * j];
    sum += v.x * v.x + v.y * v.y + v.z * v.z + v.w * v.w;
    bf16x4 h = { (__bf16)v.x, (__bf16)v.y, (__bf16)v.z, (__bf16)v.w };
    d4[lane + 32 * j] = h;
  }
#pragma unroll
  for (int off = 16; off > 0; off >>= 1)
    sum += __shfl_down(sum, off, 32);
  if (lane == 0) sq[r] = sum;
}

// -------------------------------------------------------------------------
// Async global->LDS copy of one 16B beat (ASYNCcnt-tracked DMA, no VGPR data).
// LDS byte offset = low 32 bits of the flat LDS-aperture address.
// -------------------------------------------------------------------------
__device__ __forceinline__ void async_cp16(const void* lds_dst, const void* gsrc) {
  uint32_t l = (uint32_t)(uintptr_t)lds_dst;
  uint64_t g = (uint64_t)(uintptr_t)gsrc;
  asm volatile("global_load_async_to_lds_b128 %0, %1, off"
               :: "v"(l), "v"(g) : "memory");
}
__device__ __forceinline__ void async_wait0() {
  asm volatile("s_wait_asynccnt 0" ::: "memory");
}

// -------------------------------------------------------------------------
// Pass 2: tiled bf16 WMMA GEMM + RBF epilogue.
// Workgroup: 256 threads = 8 waves (4 x 2 wave grid), 256x128 output tile.
// Each wave: 4 (M) x 4 (N) tiles of 16x16 -> 64x64, 128 f32 acc VGPRs.
// Pipeline per chunk k (program order):
//   s_wait_asynccnt 0 (chunk k DMA, issued in iter k-1) -> barrier ->
//   issue DMA chunk k+1 -> ds_load fragments(k) -> wmma(k)
// so the chunk-k WMMAs execute while chunk-(k+1) DMA is in flight.
// -------------------------------------------------------------------------
__global__ __launch_bounds__(256) void rbf_wmma(
    const bf16_t* __restrict__ abf, const bf16_t* __restrict__ bbf,
    const float* __restrict__ asq, const float* __restrict__ bsq,
    float* __restrict__ out)
{
  __shared__ __align__(16) bf16_t ldsA[2][TILE_M * LDS_STRIDE];  // 2 x 20 KB
  __shared__ __align__(16) bf16_t ldsB[2][TILE_N * LDS_STRIDE];  // 2 x 10 KB

  const int tid  = threadIdx.x;
  const int lane = tid & 31;
  const int w    = tid >> 5;      // wave 0..7
  const int wrow = w >> 1;        // 0..3  -> 64 rows each
  const int wcol = w & 1;         // 0..1  -> 64 cols each

  const int tileM = blockIdx.y * TILE_M;
  const int tileN = blockIdx.x * TILE_N;

  floatx8 acc[4][4];
#pragma unroll
  for (int mt = 0; mt < 4; ++mt)
#pragma unroll
    for (int nt = 0; nt < 4; ++nt)
      acc[mt][nt] = (floatx8)(0.0f);

  // A slab: 256x32 bf16 = 1024 x 16B beats -> 4 per thread.
  // B slab: 128x32 bf16 =  512 x 16B beats -> 2 per thread.
  auto issue_async = [&](int kc, int buf) {
#pragma unroll
    for (int i = 0; i < 4; ++i) {
      int u   = tid + 256 * i;    // 0..1023
      int row = u >> 2;
      int cb  = u & 3;
      async_cp16(&ldsA[buf][row * LDS_STRIDE + cb * 8],
                 abf + (size_t)(tileM + row) * DIM + kc + cb * 8);
    }
#pragma unroll
    for (int i = 0; i < 2; ++i) {
      int u   = tid + 256 * i;    // 0..511
      int row = u >> 2;
      int cb  = u & 3;
      async_cp16(&ldsB[buf][row * LDS_STRIDE + cb * 8],
                 bbf + (size_t)(tileN + row) * DIM + kc + cb * 8);
    }
  };

  issue_async(0, 0);              // prologue: chunk 0 -> buf 0

  const int r16 = lane & 15;
  const int hiK = (lane >> 4) * 8;   // lanes 16-31 hold K+8 (ISA 7.12.2 16-bit layout)

  const int NKC = DIM / KC;          // 16
  for (int kci = 0; kci < NKC; ++kci) {
    const int cur = kci & 1;

    async_wait0();                // chunk kci's DMA (ours) has landed in LDS
    __syncthreads();              // everyone's chunk kci landed; buf cur^1 reads done

    if (kci + 1 < NKC) issue_async((kci + 1) * KC, cur ^ 1);  // DMA next chunk

    bf16x16 afrag[4], bfrag[4];
#pragma unroll
    for (int mt = 0; mt < 4; ++mt) {
      const bf16_t* p = &ldsA[cur][(wrow * 64 + mt * 16 + r16) * LDS_STRIDE + hiK];
      bf16x8 lo = *(const bf16x8*)p;          // K = base+0..7
      bf16x8 hi = *(const bf16x8*)(p + 16);   // K = base+16..23
      afrag[mt] = __builtin_shufflevector(lo, hi, 0,1,2,3,4,5,6,7,8,9,10,11,12,13,14,15);
    }
#pragma unroll
    for (int nt = 0; nt < 4; ++nt) {
      const bf16_t* p = &ldsB[cur][(wcol * 64 + nt * 16 + r16) * LDS_STRIDE + hiK];
      bf16x8 lo = *(const bf16x8*)p;
      bf16x8 hi = *(const bf16x8*)(p + 16);
      bfrag[nt] = __builtin_shufflevector(lo, hi, 0,1,2,3,4,5,6,7,8,9,10,11,12,13,14,15);
    }

#pragma unroll
    for (int mt = 0; mt < 4; ++mt)
#pragma unroll
      for (int nt = 0; nt < 4; ++nt)
        acc[mt][nt] = __builtin_amdgcn_wmma_f32_16x16x32_bf16(
            false, afrag[mt], false, bfrag[nt],
            (short)0, acc[mt][nt], false, false);
  }

  // Epilogue: out[M,N] = exp(min(2*cross - ||a||^2 - ||b||^2, 0)), gamma = 1
#pragma unroll
  for (int nt = 0; nt < 4; ++nt) {
    const int N  = tileN + wcol * 64 + nt * 16 + r16;
    const float bn = bsq[N];
#pragma unroll
    for (int mt = 0; mt < 4; ++mt) {
      const int Mbase = tileM + wrow * 64 + mt * 16 + (lane >> 4) * 8;
#pragma unroll
      for (int v = 0; v < 8; ++v) {
        const int M = Mbase + v;
        float x = 2.0f * acc[mt][nt][v] - asq[M] - bn;
        x = fminf(x, 0.0f);                         // clamp sq_dist >= 0
        out[(size_t)M * NROWS + N] = __expf(x);
      }
    }
  }
}

// -------------------------------------------------------------------------
extern "C" void kernel_launch(void* const* d_in, const int* in_sizes, int n_in,
                              void* d_out, int out_size, void* d_ws, size_t ws_size,
                              hipStream_t stream) {
  (void)in_sizes; (void)n_in; (void)out_size; (void)ws_size;

  const float* a = (const float*)d_in[0];
  const float* b = (const float*)d_in[1];
  float* out = (float*)d_out;

  char* ws = (char*)d_ws;
  bf16_t* abf = (bf16_t*)ws;                                       //  8 MB
  bf16_t* bbf = (bf16_t*)(ws + (size_t)NROWS * DIM * 2);           //  8 MB
  float*  asq = (float*)(ws + (size_t)NROWS * DIM * 4);            // 32 KB
  float*  bsq = asq + NROWS;                                       // 32 KB

  rbf_convert<<<(2 * NROWS) / 8, 256, 0, stream>>>(a, b, abf, bbf, asq, bsq);
  rbf_wmma<<<dim3(NROWS / TILE_N, NROWS / TILE_M), 256, 0, stream>>>(abf, bbf, asq, bsq, out);
}
